// Unified_Meta_PConv_37967510896843
// MI455X (gfx1250) — matrix operand order repaired and checked by reading, not compiled, not run on hardware
//
#include <hip/hip_runtime.h>

// ---------------- types ----------------
typedef __attribute__((ext_vector_type(16))) __bf16 v16bf;
typedef __attribute__((ext_vector_type(8)))  __bf16 v8bf;
typedef __attribute__((ext_vector_type(8)))  float  v8f;
typedef __attribute__((ext_vector_type(4)))  float  v4f;

// ---------------- problem constants ----------------
#define B_      8
#define C_      256
#define T_      16384
#define O_      256
#define G_      4
#define CG      64          // channels per group
#define OG      64          // out channels per group
#define TI      8192        // per-phase time length
#define TILE_N  128         // time columns per block
#define XS_ROWS 136         // staged xs rows: global t in [tile_start-4, tile_start+131]
#define XSF_STRIDE 68       // fp32 per row (64 + 4 pad -> conflict-free frag reads)
#define T0_STRIDE 136       // t0buf row stride (fp32)
#define KSTEPS  10          // K = 320 = 64ch * 5 taps, 32 per WMMA
#define MTILES  12          // M = 192 = 3 outputs * 64 oc, 16 per WMMA
#define BOUNDARY_F 1.41421356237309515f

// =====================================================================
// Prep: build effective weights, packed directly in WMMA A-fragment
// per-lane layout: frag[((g*MTILES+mt)*KSTEPS+kk)*32 + lane][16 bf16]
// A k order used here (contiguous, same as B): lane<16 -> M=lane, slots 0..15 = K 0..15
//                      lane>=16 -> M=lane-16, slots 0..15 = K 16..31
// K index = j*64 + c  (j = tap 0..4, c = channel 0..63)
// M row: [0,64) t0 kernel, [64,128) d1 kernel, [128,192) d2 kernel
// =====================================================================
__global__ __launch_bounds__(256) void weff_prep(const float* __restrict__ w,
                                                 __bf16* __restrict__ frag) {
  int idx = blockIdx.x * 256 + threadIdx.x;
  if (idx >= G_ * MTILES * KSTEPS * 32) return;
  int lane = idx & 31;
  int kk   = (idx >> 5) % KSTEPS;
  int mt   = (idx / (32 * KSTEPS)) % MTILES;
  int g    = idx / (32 * KSTEPS * MTILES);

  int Mrow    = mt * 16 + (lane & 15);
  int sel     = Mrow >> 6;        // 0:t0  1:d1  2:d2
  int o_local = Mrow & 63;
  int o_glob  = g * OG + o_local;

  __bf16* dst = frag + (size_t)idx * 16;
  for (int s = 0; s < 16; ++s) {
    int klocal = (lane < 16) ? s : s + 16;
    int K = kk * 32 + klocal;     // 0..319
    int j = K >> 6;               // tap 0..4
    int c = K & 63;               // channel
    const float* wp = w + ((size_t)o_glob * CG + c) * 13;
    float val;
    if (sel == 0) {
      val = wp[3 * j];                                   // taps 0,3,6,9,12
    } else {
      int base = (sel == 1) ? 2 : 1;                     // d1: k%3==2, d2: k%3==1
      float a  = (j >= 1) ? wp[3 * (j - 1) + base] : 0.f;
      float bb = (j <= 3) ? wp[3 * j + base]       : 0.f;
      val = a - bb;                                      // Kd[m] = w[m-1] - w[m]
    }
    dst[s] = (__bf16)val;
  }
}

// =====================================================================
// Main: per block = (batch b, phase ph, group g, time tile).
// xs tile staged memory->LDS with global_load_async_to_lds_b32 (ASYNCcnt),
// GEMM M=192 N=128 K=320 with v_wmma_f32_16x16x32_bf16, then pointwise
// denom/combine with a 4-column t0 halo computed by VALU.
// =====================================================================
__global__ __launch_bounds__(256) void pconv_main(const float*  __restrict__ x,
                                                  const float*  __restrict__ w,
                                                  const float*  __restrict__ bias,
                                                  const __bf16* __restrict__ frag,
                                                  float*        __restrict__ out) {
  __shared__ float xsf[XS_ROWS * XSF_STRIDE];    // ~37.0 KB, fp32 (async DMA target)
  __shared__ float t0buf[OG * T0_STRIDE];        // ~34.8 KB

  int bid  = blockIdx.x;
  int tile = bid & 63;
  int g    = (bid >> 6) & 3;
  int ph   = (bid >> 8) & 1;
  int b    = bid >> 9;

  int tid  = threadIdx.x;
  int lane = tid & 31;
  int wave = tid >> 5;

  int tile_start = tile * TILE_N;

  // ---- zero-fill the 8 potential OOB halo rows (rows 0..3 and 132..135) ----
  for (int k = tid; k < 8 * CG; k += 256) {
    int rr = k >> 6;                       // 0..7
    int c  = k & 63;
    int r  = (rr < 4) ? rr : (128 + rr);   // 0..3, 132..135
    xsf[r * XSF_STRIDE + c] = 0.f;
  }
  __syncthreads();   // order DS zero-fill before async DMA writes

  // ---- stage xs tile (+/-4 halo) into LDS with async loads ----
  const float* xg = x + ((size_t)b * C_ + (size_t)g * CG) * T_;
  for (int it = 0; it < (XS_ROWS * CG) / 256; ++it) {
    int idx = it * 256 + tid;
    int r = idx % XS_ROWS;        // consecutive tid -> consecutive time
    int c = idx / XS_ROWS;
    int t = tile_start - 4 + r;   // xs index (pre-dilation-sampling)
    if (t >= 0 && t < TI) {
      unsigned voff = (unsigned)((c * T_ + 2 * t + ph) * 4);       // byte offset
      unsigned ldsa = (unsigned)(size_t)(&xsf[r * XSF_STRIDE + c]); // LDS byte addr
      asm volatile("global_load_async_to_lds_b32 %0, %1, %2"
                   :: "v"(ldsa), "v"(voff), "s"(xg)
                   : "memory");
    }
  }
  asm volatile("s_wait_asynccnt 0x0" ::: "memory");
  __syncthreads();

  // ---- WMMA main loop: each wave owns a 192x16 column strip ----
  v8f acc[MTILES];
  v8f zero = {0.f, 0.f, 0.f, 0.f, 0.f, 0.f, 0.f, 0.f};
#pragma unroll
  for (int m = 0; m < MTILES; ++m) acc[m] = zero;

  const __bf16* fragG = frag + (size_t)g * MTILES * KSTEPS * 32 * 16;
  int nloc = wave * 16 + (lane & 15);   // local output column of this lane

  for (int kk = 0; kk < KSTEPS; ++kk) {
    // B fragment 32x16: lane<16 holds K=Kb..Kb+15 for column nloc (contiguous run
    // stays within one tap j since K = j*64 + c and runs are 16-aligned)
    int Kb = kk * 32 + ((lane >= 16) ? 16 : 0);
    int j  = Kb >> 6;
    int c0 = Kb & 63;
    const float* src = &xsf[(size_t)(nloc + j + 2) * XSF_STRIDE + c0];
    v4f f0 = *(const v4f*)(src);
    v4f f1 = *(const v4f*)(src + 4);
    v4f f2 = *(const v4f*)(src + 8);
    v4f f3 = *(const v4f*)(src + 12);
    union { v16bf v; __bf16 e[16]; } bf;
#pragma unroll
    for (int q = 0; q < 4; ++q) bf.e[q]      = (__bf16)f0[q];
#pragma unroll
    for (int q = 0; q < 4; ++q) bf.e[4 + q]  = (__bf16)f1[q];
#pragma unroll
    for (int q = 0; q < 4; ++q) bf.e[8 + q]  = (__bf16)f2[q];
#pragma unroll
    for (int q = 0; q < 4; ++q) bf.e[12 + q] = (__bf16)f3[q];

#pragma unroll
    for (int m = 0; m < MTILES; ++m) {
      const __bf16* ap = fragG + (((size_t)m * KSTEPS + kk) * 32 + lane) * 16;
      union { v16bf v; v8bf h[2]; } af;
      af.h[0] = *(const v8bf*)(ap);
      af.h[1] = *(const v8bf*)(ap + 8);
      acc[m] = __builtin_amdgcn_wmma_f32_16x16x32_bf16(
          false, af.v, false, bf.v, (short)0, acc[m], false, false);
    }
  }

  // ---- halo t0 columns (n_local in {-2,-1,128,129}); one value/thread ----
  {
    int h = tid >> 6;                       // 0..3
    int o_local = tid & 63;
    int nh = (h < 2) ? (h - 2) : (126 + h); // -2,-1,128,129
    int tt = tile_start + nh;
    float v = 0.f;
    if (tt >= 2 && tt < TI - 2) {
      v = bias[g * OG + o_local];
      const float* wp = w + (size_t)(g * OG + o_local) * CG * 13;
      for (int j = 0; j < 5; ++j) {
        int r = nh + j + 2;
        for (int c = 0; c < CG; ++c)
          v += wp[c * 13 + 3 * j] * xsf[r * XSF_STRIDE + c];
      }
    }
    t0buf[o_local * T0_STRIDE + (nh + 2)] = v;
  }

  // ---- write main-tile t0 (+bias, border-masked) to LDS ----
  {
    int tt = tile_start + nloc;
    bool inr = (tt >= 2 && tt < TI - 2);
#pragma unroll
    for (int m = 0; m < 4; ++m) {
#pragma unroll
      for (int v = 0; v < 8; ++v) {
        int orow = m * 16 + v + ((lane >= 16) ? 8 : 0);  // C/D layout
        float t0v = inr ? (acc[m][v] + bias[g * OG + orow]) : 0.f;
        t0buf[orow * T0_STRIDE + (nloc + 2)] = t0v;
      }
    }
  }
  __syncthreads();

  // ---- combine: denom from t0 window, mix t0/d1/d2, scatter out ----
  {
    int tt = tile_start + nloc;
    bool inr = (tt >= 2 && tt < TI - 2);
    float* outp = out + ((size_t)b * O_ + (size_t)g * OG) * T_;
#pragma unroll
    for (int m = 0; m < 4; ++m) {
#pragma unroll
      for (int v = 0; v < 8; ++v) {
        int orow = m * 16 + v + ((lane >= 16) ? 8 : 0);
        const float* tb = &t0buf[orow * T0_STRIDE + (nloc + 2)];
        float t0c = tb[0];
        float s5  = tb[-2] + tb[-1] + t0c + tb[1] + tb[2];
        float den = t0c - 0.2f * s5;
        den = fminf(fmaxf(den, 1.0f), BOUNDARY_F);
        float inv = 1.0f / den;
        float d1 = inr ? acc[4 + m][v] : 0.f;
        float d2 = inr ? acc[8 + m][v] : 0.f;
        float res = 0.25f * (t0c + (d1 + d2) * inv + 2.0f * (d2 - d1) * inv * inv);
        outp[(size_t)orow * T_ + 2 * tt + ph] = res;
      }
    }
  }
}

// =====================================================================
extern "C" void kernel_launch(void* const* d_in, const int* in_sizes, int n_in,
                              void* d_out, int out_size, void* d_ws, size_t ws_size,
                              hipStream_t stream) {
  (void)in_sizes; (void)n_in; (void)out_size; (void)ws_size;
  const float* x = (const float*)d_in[0];
  const float* w = (const float*)d_in[1];
  const float* b = (const float*)d_in[2];
  float* out = (float*)d_out;
  __bf16* frag = (__bf16*)d_ws;   // 491,520 B of packed effective weights

  int prep_threads = G_ * MTILES * KSTEPS * 32;   // 15360
  weff_prep<<<(prep_threads + 255) / 256, 256, 0, stream>>>(w, frag);

  int nblocks = B_ * 2 * G_ * (TI / TILE_N);      // 8*2*4*64 = 4096
  pconv_main<<<nblocks, 256, 0, stream>>>(x, w, b, frag, out);
}
